// LinearKernelAttention_23081154248844
// MI455X (gfx1250) — matrix operand order repaired and checked by reading, not compiled, not run on hardware
//
#include <hip/hip_runtime.h>

typedef __attribute__((ext_vector_type(16))) _Float16 v16h;
typedef __attribute__((ext_vector_type(8)))  _Float16 v8h;
typedef __attribute__((ext_vector_type(8)))  float    v8f;

#define E_DIM   128
#define NHEAD   4
#define DHEAD   32
#define CTOK    2048
#define CHUNK   128
#define NCHUNK  (CTOK / CHUNK)

// ---- LDS layout (dynamic shared) ----
// sW0sw : 128*128 f16 fragment-swizzled (Wk p1 / Wq p2)   off 0       32768
// sW1sw : 128*128 f16 fragment-swizzled (Wv p1 / Wo p2)   off 32768   32768
// sX    : 2 x 128*128 f16 row-major input (double buf)    off 65536   65536
// sS0   : 128*128 f16 (K col-major p1 / Q row-major p2)   off 131072  32768
// sS1   : 128*128 f16 (V col-major p1 / O row-major p2)   off 163840  32768
//   -> sS0+sS1 doubles as 64KB f32 TDM weight scratch (off 131072)
// sKtVrm: 4*32*32 f16 row-major                           off 196608  8192
// sKtVsw: 8 frags * 32 lanes * 16 f16                     off 204800  8192
// sGsw  : 4 frags * 32 lanes * 16 f16 (Z matrix)          off 212992  4096
// sKsum : 128 f32                                         off 217088  512
// sZ    : 128*4 f32                                       off 217600  2048
// sB    : 4*128 f32 (bq,bk,bv,bo)                         off 219648  2048
#define SMEM_BYTES     221696
#define WSCRATCH_OFF   131072u
#define XBUF_HALVES    16384   // 32KB per buffer

__device__ __forceinline__ v8f wmma_f16(v16h a, v16h b, v8f c) {
  return __builtin_amdgcn_wmma_f32_16x16x32_f16(false, a, false, b, (short)0, c, false, false);
}

// A fragment (16x32 MxK) from row-major f16 matrix: one contiguous 32B run (k = 16h + s).
__device__ __forceinline__ v16h load_a(const _Float16* M, int ldm, int m0, int k0, int lane) {
  const _Float16* p = M + (m0 + (lane & 15)) * ldm + k0 + ((lane & 16) ? 16 : 0);
  v8h lo = *(const v8h*)p;
  v8h hi = *(const v8h*)(p + 8);
  return __builtin_shufflevector(lo, hi, 0,1,2,3,4,5,6,7,8,9,10,11,12,13,14,15);
}

// B fragment from fragment-swizzled storage: [frag][lane][16 halves] contiguous.
__device__ __forceinline__ v16h load_bsw(const _Float16* base, int frag, int lane) {
  const _Float16* p = base + ((frag * 32 + lane) << 4);
  v8h lo = *(const v8h*)p;
  v8h hi = *(const v8h*)(p + 8);
  return __builtin_shufflevector(lo, hi, 0,1,2,3,4,5,6,7,8,9,10,11,12,13,14,15);
}

// B fragment (32x16 KxN) with logical B[k][n] = M[n*ldm + k] (column-major src).
__device__ __forceinline__ v16h load_b_t(const _Float16* M, int ldm, int k0, int n0, int lane) {
  const _Float16* p = M + (n0 + (lane & 15)) * ldm + k0 + ((lane & 16) ? 16 : 0);
  v8h lo = *(const v8h*)p;
  v8h hi = *(const v8h*)(p + 8);
  return __builtin_shufflevector(lo, hi, 0,1,2,3,4,5,6,7,8,9,10,11,12,13,14,15);
}

__device__ __forceinline__ float elu1(float x) {
  return x > 0.0f ? x + 1.0f : __expf(x);
}

// TDM: DMA 64KB (8192 x 8B) from global to LDS at byte offset lds_off.
// D# packed per CDNA5 ISA 8.3/8.4. Issue from one wave only; EXEC ignored.
__device__ __forceinline__ void tdm_load_64kb(const void* gsrc, unsigned lds_off) {
  typedef __attribute__((ext_vector_type(4))) unsigned u32x4;
  typedef __attribute__((ext_vector_type(8))) int      i32x8;
  typedef __attribute__((ext_vector_type(4))) int      i32x4;
  unsigned long long ga = (unsigned long long)gsrc;
  u32x4 g0;
  g0[0] = 1u;                                            // count=1, user mode
  g0[1] = lds_off;                                       // lds_addr
  g0[2] = (unsigned)(ga & 0xFFFFFFFFull);                // global_addr[31:0]
  g0[3] = (unsigned)((ga >> 32) & 0x01FFFFFFull)         // global_addr[56:32]
        | 0x80000000u;                                   // type=2 (bits 127:126)
  i32x8 g1;
  g1[0] = 0x00030000;   // wg_mask=0, data_size=3 (8B)
  g1[1] = 0x20000000;   // tensor_dim0[15:0]=8192 in bits[63:48]
  g1[2] = 0x00010000;   // tensor_dim0[31:16]=0, tensor_dim1[15:0]=1
  g1[3] = 0x20000000;   // tensor_dim1[31:16]=0, tile_dim0=8192
  g1[4] = 1;            // tile_dim1=1, tile_dim2=0
  g1[5] = 8192;         // tensor_dim0_stride[31:0]
  g1[6] = 0;
  g1[7] = 0;
  i32x4 g2 = {0, 0, 0, 0};
  i32x4 g3 = {0, 0, 0, 0};
#if defined(__clang_major__) && __clang_major__ >= 23
  i32x8 g4 = {0, 0, 0, 0, 0, 0, 0, 0};
  __builtin_amdgcn_tensor_load_to_lds(g0, g1, g2, g3, g4, 0);
#else
  __builtin_amdgcn_tensor_load_to_lds(g0, g1, g2, g3, 0);
#endif
}

// Swizzle a 128x128 f32 weight (KxN row-major, staged in LDS) into
// B-fragment-major f16: frag = nt*4 + kt.
__device__ __forceinline__ void build_wsw(const float* __restrict__ Wsrc, _Float16* dst, int tid) {
  for (int idx = tid; idx < E_DIM * E_DIM; idx += 256) {
    int e    = idx & 15;
    int ln   = (idx >> 4) & 31;
    int frag = idx >> 9;
    int kt   = frag & 3;
    int nt   = frag >> 2;
    int k = kt * 32 + ((ln & 16) ? 16 : 0) + e;
    int n = nt * 16 + (ln & 15);
    dst[idx] = (_Float16)Wsrc[k * E_DIM + n];
  }
}

// 2x4 register-blocked 128x128 @ 128x128 projection.
template<bool ELU, bool COLMAJOR>
__device__ __forceinline__ void projection(const _Float16* sXa, const _Float16* sWsw,
                                           _Float16* sDst, const float* bias,
                                           int wave, int lane) {
  const int mt2 = wave >> 1;
  const int nh  = wave & 1;
  v8f acc[2][4];
#pragma unroll
  for (int im = 0; im < 2; ++im)
#pragma unroll
    for (int nt = 0; nt < 4; ++nt) { v8f z = {}; acc[im][nt] = z; }

#pragma unroll
  for (int kt = 0; kt < 4; ++kt) {
    v16h a0 = load_a(sXa, E_DIM, mt2 * 32 + 0,  kt * 32, lane);
    v16h a1 = load_a(sXa, E_DIM, mt2 * 32 + 16, kt * 32, lane);
#pragma unroll
    for (int nt = 0; nt < 4; ++nt) {
      v16h b = load_bsw(sWsw, (nh * 4 + nt) * 4 + kt, lane);
      acc[0][nt] = wmma_f16(a0, b, acc[0][nt]);
      acc[1][nt] = wmma_f16(a1, b, acc[1][nt]);
    }
  }
#pragma unroll
  for (int im = 0; im < 2; ++im) {
    int mrow = mt2 * 32 + im * 16 + ((lane & 16) ? 8 : 0);
#pragma unroll
    for (int nt = 0; nt < 4; ++nt) {
      int n = nh * 64 + nt * 16 + (lane & 15);
      float bs = bias[n];
      if (COLMAJOR) {
        v8h pk;
#pragma unroll
        for (int i = 0; i < 8; ++i) {
          float v = acc[im][nt][i] + bs;
          if (ELU) v = elu1(v);
          pk[i] = (_Float16)v;
        }
        *(v8h*)(sDst + n * E_DIM + mrow) = pk;
      } else {
#pragma unroll
        for (int i = 0; i < 8; ++i) {
          float v = acc[im][nt][i] + bs;
          if (ELU) v = elu1(v);
          sDst[(mrow + i) * E_DIM + n] = (_Float16)v;
        }
      }
    }
  }
}

// Direct stage: f32 global -> f16 LDS (used only for chunk 0 of each pass).
__device__ __forceinline__ void stage_direct(const float* gsrc, _Float16* dst, int tid) {
  const float4* xg = (const float4*)gsrc;
  for (int i = tid; i < CHUNK * E_DIM / 8; i += 256) {
    float4 t0 = xg[2 * i];
    float4 t1 = xg[2 * i + 1];
    v8h h;
    h[0] = (_Float16)t0.x; h[1] = (_Float16)t0.y; h[2] = (_Float16)t0.z; h[3] = (_Float16)t0.w;
    h[4] = (_Float16)t1.x; h[5] = (_Float16)t1.y; h[6] = (_Float16)t1.z; h[7] = (_Float16)t1.w;
    *(v8h*)(dst + 8 * i) = h;
  }
}

__global__ __launch_bounds__(256, 1)
void linattn_fused_kernel(const float* __restrict__ X,
                          const float* __restrict__ Wq, const float* __restrict__ bq,
                          const float* __restrict__ Wk, const float* __restrict__ bk,
                          const float* __restrict__ Wv, const float* __restrict__ bv,
                          const float* __restrict__ Wo, const float* __restrict__ bo,
                          float* __restrict__ Out) {
  extern __shared__ char smem[];
  _Float16* sW0sw = (_Float16*)(smem);
  _Float16* sW1sw = (_Float16*)(smem + 32768);
  _Float16* sX0   = (_Float16*)(smem + 65536);   // double buffer: +0 / +XBUF_HALVES
  _Float16* sS0   = (_Float16*)(smem + 131072);
  _Float16* sS1   = (_Float16*)(smem + 163840);
  const float* sWscratch = (const float*)(smem + WSCRATCH_OFF);
  _Float16* sKtVrm= (_Float16*)(smem + 196608);
  _Float16* sKtVsw= (_Float16*)(smem + 204800);
  _Float16* sGsw  = (_Float16*)(smem + 212992);
  float*    sKsum = (float*)   (smem + 217088);
  float*    sZ    = (float*)   (smem + 217600);
  float*    sB    = (float*)   (smem + 219648);

  const int tid  = threadIdx.x;
  const int wave = tid >> 5;
  const int lane = tid & 31;
  const int head = wave >> 1;
  const int trow = wave & 1;
  const long long seq_base = (long long)blockIdx.x * (long long)(CTOK * E_DIM);

  if (tid < 128) {
    sB[tid]       = bq[tid];
    sB[128 + tid] = bk[tid];
    sB[256 + tid] = bv[tid];
    sB[384 + tid] = bo[tid];
    sKsum[tid]    = 0.0f;
  }

  // ---- pass-1 weights via Tensor Data Mover: Wk, Wv -> scratch -> swizzle ----
  if (wave == 0) {
    tdm_load_64kb(Wk, WSCRATCH_OFF);
    __builtin_amdgcn_s_wait_tensorcnt(0);
  }
  __syncthreads();
  build_wsw(sWscratch, sW0sw, tid);
  __syncthreads();
  if (wave == 0) {
    tdm_load_64kb(Wv, WSCRATCH_OFF);
    __builtin_amdgcn_s_wait_tensorcnt(0);
  }
  __syncthreads();
  build_wsw(sWscratch, sW1sw, tid);

  // stage chunk 0 into buffer 0
  stage_direct(X + seq_base, sX0, tid);
  __syncthreads();

  v8f ktv0 = {};
  v8f ktv1 = {};

  // ======================= PASS 1: K,V -> KtV, ksum =======================
  for (int ch = 0; ch < NCHUNK; ++ch) {
    const int sel = ch & 1;
    const _Float16* cur = sX0 + sel * XBUF_HALVES;          // offset select keeps LDS AS
    _Float16*       nxt = sX0 + (sel ^ 1) * XBUF_HALVES;
    const bool havePf = (ch + 1 < NCHUNK);

    // issue next chunk's global loads early; consumed after the GEMMs
    float4 pf[16];
    const float4* xn = (const float4*)(X + seq_base + (long long)(ch + 1) * CHUNK * E_DIM);
    if (havePf) {
#pragma unroll
      for (int j = 0; j < 8; ++j) {
        pf[2 * j]     = xn[2 * (tid + j * 256)];
        pf[2 * j + 1] = xn[2 * (tid + j * 256) + 1];
      }
    }

    // K = elu1(X@Wk + bk) col-major; V = X@Wv + bv col-major
    projection<true,  true>(cur, sW0sw, sS0, sB + 128, wave, lane);
    projection<false, true>(cur, sW1sw, sS1, sB + 256, wave, lane);
    __syncthreads();

    // ksum[dim] += sum over tokens (contiguous in column-major K)
    if (tid < 128) {
      const _Float16* col = sS0 + tid * E_DIM;
      float s = 0.0f;
#pragma unroll
      for (int v = 0; v < 16; ++v) {
        v8h h = *(const v8h*)(col + 8 * v);
#pragma unroll
        for (int i = 0; i < 8; ++i) s += (float)h[i];
      }
      sKsum[tid] += s;
    }

    // KtV_h += K_h^T @ V_h
#pragma unroll
    for (int kb = 0; kb < 4; ++kb) {
      v16h a  = load_a(sS0, E_DIM, head * 32 + trow * 16, kb * 32, lane);
      v16h b0 = load_b_t(sS1, E_DIM, kb * 32, head * 32 + 0,  lane);
      v16h b1 = load_b_t(sS1, E_DIM, kb * 32, head * 32 + 16, lane);
      ktv0 = wmma_f16(a, b0, ktv0);
      ktv1 = wmma_f16(a, b1, ktv1);
    }

    // commit prefetched chunk into the alternate buffer
    if (havePf) {
#pragma unroll
      for (int j = 0; j < 8; ++j) {
        v8h h;
        h[0] = (_Float16)pf[2*j].x;   h[1] = (_Float16)pf[2*j].y;
        h[2] = (_Float16)pf[2*j].z;   h[3] = (_Float16)pf[2*j].w;
        h[4] = (_Float16)pf[2*j+1].x; h[5] = (_Float16)pf[2*j+1].y;
        h[6] = (_Float16)pf[2*j+1].z; h[7] = (_Float16)pf[2*j+1].w;
        *(v8h*)(nxt + 8 * (tid + j * 256)) = h;
      }
    }
    __syncthreads();
  }

  // publish KtV row-major
  {
    int m = trow * 16 + ((lane & 16) ? 8 : 0);
#pragma unroll
    for (int i = 0; i < 8; ++i) {
      sKtVrm[head * 1024 + (m + i) * 32 + 0  + (lane & 15)] = (_Float16)ktv0[i];
      sKtVrm[head * 1024 + (m + i) * 32 + 16 + (lane & 15)] = (_Float16)ktv1[i];
    }
  }
  __syncthreads();

  // fragment-swizzled KtV (frag = h*2 + nt) ; TDM fetch Wq in parallel
  if (wave == 0) {
    tdm_load_64kb(Wq, WSCRATCH_OFF);
    __builtin_amdgcn_s_wait_tensorcnt(0);
  }
  for (int idx = tid; idx < 8 * 32 * 16; idx += 256) {
    int e    = idx & 15;
    int ln   = (idx >> 4) & 31;
    int frag = idx >> 9;
    int nt   = frag & 1;
    int h    = frag >> 1;
    int k = ((ln & 16) ? 16 : 0) + e;
    int n = nt * 16 + (ln & 15);
    sKtVsw[idx] = sKtVrm[h * 1024 + k * 32 + n];
  }
  // fragment-swizzled Z matrix G (128x16): G[d][h] = ksum[d] iff d/32==h
  for (int idx = tid; idx < 4 * 32 * 16; idx += 256) {
    int e  = idx & 15;
    int ln = (idx >> 4) & 31;
    int kt = idx >> 9;
    int k = kt * 32 + ((ln & 16) ? 16 : 0) + e;
    int n = ln & 15;
    sGsw[idx] = (n == (k >> 5)) ? (_Float16)sKsum[k] : (_Float16)0.0f;
  }
  __syncthreads();
  build_wsw(sWscratch, sW0sw, tid);   // Wq
  __syncthreads();
  if (wave == 0) {
    tdm_load_64kb(Wo, WSCRATCH_OFF);
    __builtin_amdgcn_s_wait_tensorcnt(0);
  }
  __syncthreads();
  build_wsw(sWscratch, sW1sw, tid);   // Wo
  stage_direct(X + seq_base, sX0, tid);
  __syncthreads();

  // ======================= PASS 2: Q -> out -> Wo ========================
  for (int ch = 0; ch < NCHUNK; ++ch) {
    const int sel = ch & 1;
    const _Float16* cur = sX0 + sel * XBUF_HALVES;
    _Float16*       nxt = sX0 + (sel ^ 1) * XBUF_HALVES;
    const bool havePf = (ch + 1 < NCHUNK);

    float4 pf[16];
    const float4* xn = (const float4*)(X + seq_base + (long long)(ch + 1) * CHUNK * E_DIM);
    if (havePf) {
#pragma unroll
      for (int j = 0; j < 8; ++j) {
        pf[2 * j]     = xn[2 * (tid + j * 256)];
        pf[2 * j + 1] = xn[2 * (tid + j * 256) + 1];
      }
    }

    // Q = elu1(X@Wq + bq) row-major
    projection<true, false>(cur, sW0sw, sS0, sB, wave, lane);
    __syncthreads();

    // Z via WMMA: denom(128x16) = Q @ G ; wave w -> tokens 16w
    {
      v8f c = {};
#pragma unroll
      for (int kt = 0; kt < 4; ++kt) {
        v16h a = load_a(sS0, E_DIM, wave * 16, kt * 32, lane);
        v16h b = load_bsw(sGsw, kt, lane);
        c = wmma_f16(a, b, c);
      }
      int h = lane & 15;
      if (h < NHEAD) {
        int mrow = wave * 16 + ((lane & 16) ? 8 : 0);
#pragma unroll
        for (int i = 0; i < 8; ++i)
          sZ[(mrow + i) * 4 + h] = 1.0f / (c[i] + 1e-6f);
      }
    }
    __syncthreads();

    // out_h = (Q_h @ KtV_h) * Z -> sS1 row-major
#pragma unroll
    for (int h = 0; h < NHEAD; ++h) {
      v16h a = load_a(sS0, E_DIM, wave * 16, h * 32, lane);
#pragma unroll
      for (int nt = 0; nt < 2; ++nt) {
        v16h b = load_bsw(sKtVsw, h * 2 + nt, lane);
        v8f c = {};
        c = wmma_f16(a, b, c);
        int n    = h * 32 + nt * 16 + (lane & 15);
        int mrow = wave * 16 + ((lane & 16) ? 8 : 0);
#pragma unroll
        for (int i = 0; i < 8; ++i) {
          int tok = mrow + i;
          sS1[tok * E_DIM + n] = (_Float16)(c[i] * sZ[tok * 4 + h]);
        }
      }
    }
    __syncthreads();

    // final = O @ Wo + bo -> global f32 (2x4 blocked)
    {
      const int mt2 = wave >> 1;
      const int nh  = wave & 1;
      v8f acc[2][4];
#pragma unroll
      for (int im = 0; im < 2; ++im)
#pragma unroll
        for (int nt = 0; nt < 4; ++nt) { v8f z = {}; acc[im][nt] = z; }
#pragma unroll
      for (int kt = 0; kt < 4; ++kt) {
        v16h a0 = load_a(sS1, E_DIM, mt2 * 32 + 0,  kt * 32, lane);
        v16h a1 = load_a(sS1, E_DIM, mt2 * 32 + 16, kt * 32, lane);
#pragma unroll
        for (int nt = 0; nt < 4; ++nt) {
          v16h b = load_bsw(sW1sw, (nh * 4 + nt) * 4 + kt, lane);
          acc[0][nt] = wmma_f16(a0, b, acc[0][nt]);
          acc[1][nt] = wmma_f16(a1, b, acc[1][nt]);
        }
      }
      float* outp = Out + seq_base + (long long)ch * CHUNK * E_DIM;
#pragma unroll
      for (int im = 0; im < 2; ++im) {
        int mrow = mt2 * 32 + im * 16 + ((lane & 16) ? 8 : 0);
#pragma unroll
        for (int nt = 0; nt < 4; ++nt) {
          int n = nh * 64 + nt * 16 + (lane & 15);
          float bs = sB[384 + n];
#pragma unroll
          for (int i = 0; i < 8; ++i)
            outp[(mrow + i) * E_DIM + n] = acc[im][nt][i] + bs;
        }
      }
    }

    if (havePf) {
#pragma unroll
      for (int j = 0; j < 8; ++j) {
        v8h h;
        h[0] = (_Float16)pf[2*j].x;   h[1] = (_Float16)pf[2*j].y;
        h[2] = (_Float16)pf[2*j].z;   h[3] = (_Float16)pf[2*j].w;
        h[4] = (_Float16)pf[2*j+1].x; h[5] = (_Float16)pf[2*j+1].y;
        h[6] = (_Float16)pf[2*j+1].z; h[7] = (_Float16)pf[2*j+1].w;
        *(v8h*)(nxt + 8 * (tid + j * 256)) = h;
      }
    }
    __syncthreads();
  }
}

extern "C" void kernel_launch(void* const* d_in, const int* in_sizes, int n_in,
                              void* d_out, int out_size, void* d_ws, size_t ws_size,
                              hipStream_t stream) {
  (void)n_in; (void)out_size; (void)d_ws; (void)ws_size;
  const float* X  = (const float*)d_in[0];
  const float* Wq = (const float*)d_in[1];
  const float* bq = (const float*)d_in[2];
  const float* Wk = (const float*)d_in[3];
  const float* bk = (const float*)d_in[4];
  const float* Wv = (const float*)d_in[5];
  const float* bv = (const float*)d_in[6];
  const float* Wo = (const float*)d_in[7];
  const float* bo = (const float*)d_in[8];
  float* Out = (float*)d_out;

  int nseq = in_sizes[0] / (CTOK * E_DIM); // B*R = 128

  static bool attr_set = false;
  if (!attr_set) {
    hipFuncSetAttribute((const void*)linattn_fused_kernel,
                        hipFuncAttributeMaxDynamicSharedMemorySize, SMEM_BYTES);
    attr_set = true;
  }

  hipLaunchKernelGGL(linattn_fused_kernel, dim3(nseq), dim3(256), SMEM_BYTES, stream,
                     X, Wq, bq, Wk, bk, Wv, bv, Wo, bo, Out);
}
